// PostProcessor_22660247454147
// MI455X (gfx1250) — matrix operand order repaired and checked by reading, not compiled, not run on hardware
//
#include <hip/hip_runtime.h>
#include <hip/hip_bf16.h>

#define N_PROP 1000
#define N_CLS  81
#define N_C1   80          // classes minus background
#define D_FEAT 1024
#define DETS   100
#define NEG_INF (-__builtin_inff())
#define BBOX_CLIP 4.135166556742356f   // log(1000/16)

typedef __attribute__((ext_vector_type(2)))  float    v2f;
typedef __attribute__((ext_vector_type(8)))  float    v8f;
typedef __attribute__((ext_vector_type(16))) _Float16 v16h;

// ---------------------------------------------------------------------------
// K1: softmax over classes + box decode/clip.  1 thread per proposal.
// scores_c: [80][1000]  (class-major, background dropped)
// boxes_c : [80][1000][4]
// ---------------------------------------------------------------------------
__global__ __launch_bounds__(128)
void k_softmax_decode(const float* __restrict__ logits,
                      const float* __restrict__ reg,
                      const float* __restrict__ props,
                      float* __restrict__ scores_c,
                      float* __restrict__ boxes_c) {
  int n = blockIdx.x * blockDim.x + threadIdx.x;
  if (n >= N_PROP) return;
  const float* lrow = logits + n * N_CLS;
  float mx = lrow[0];
  for (int c = 1; c < N_CLS; ++c) mx = fmaxf(mx, lrow[c]);
  float ssum = 0.f;
  for (int c = 0; c < N_CLS; ++c) ssum += expf(lrow[c] - mx);
  float inv = 1.f / ssum;

  float px1 = props[n*4+0], py1 = props[n*4+1];
  float px2 = props[n*4+2], py2 = props[n*4+3];
  float w  = px2 - px1 + 1.f, h = py2 - py1 + 1.f;
  float cx = px1 + 0.5f*w,  cy = py1 + 0.5f*h;
  const float* rrow = reg + n * (N_CLS * 4);

  for (int c = 1; c < N_CLS; ++c) {
    int cc = c - 1;
    scores_c[cc*N_PROP + n] = expf(lrow[c] - mx) * inv;
    float dx = rrow[4*c+0] * 0.1f;
    float dy = rrow[4*c+1] * 0.1f;
    float dw = fminf(rrow[4*c+2] * 0.2f, BBOX_CLIP);
    float dh = fminf(rrow[4*c+3] * 0.2f, BBOX_CLIP);
    float pcx = dx*w + cx,  pcy = dy*h + cy;
    float pw  = expf(dw)*w, ph  = expf(dh)*h;
    float ox1 = fminf(fmaxf(pcx - 0.5f*pw,       0.f), 1332.f);
    float oy1 = fminf(fmaxf(pcy - 0.5f*ph,       0.f),  799.f);
    float ox2 = fminf(fmaxf(pcx + 0.5f*pw - 1.f, 0.f), 1332.f);
    float oy2 = fminf(fmaxf(pcy + 0.5f*ph - 1.f, 0.f),  799.f);
    float* bo = boxes_c + (cc*N_PROP + n)*4;
    bo[0]=ox1; bo[1]=oy1; bo[2]=ox2; bo[3]=oy2;
  }
}

// ---------------------------------------------------------------------------
// K2: per-class stable descending sort via rank counting (matches jnp.argsort
// tie-breaking).  One block per class; scores staged in LDS.
// sv[c][r] = sorted (thresholded) score, order[c][r] = original proposal idx.
// ---------------------------------------------------------------------------
__global__ __launch_bounds__(256)
void k_sort(const float* __restrict__ scores_c,
            float* __restrict__ sv, int* __restrict__ order) {
  int c = blockIdx.x, tid = threadIdx.x;
  __shared__ float s[N_PROP];
  for (int i = tid; i < N_PROP; i += 256) {
    float v = scores_c[c*N_PROP + i];
    s[i] = (v > 0.05f) ? v : NEG_INF;
  }
  __syncthreads();
  for (int i = tid; i < N_PROP; i += 256) {
    float si = s[i];
    int r = 0;
    for (int j = 0; j < N_PROP; ++j) {
      float sj = s[j];
      r += (int)((sj > si) || (sj == si && j < i));
    }
    order[c*N_PROP + r] = i;
    sv[c*N_PROP + r] = si;
  }
}

// ---------------------------------------------------------------------------
// K3: greedy NMS per class via suppression bitmask.  One block per class.
// Suppression rows are keep-independent -> computed fully parallel in 256-row
// chunks; a single wave32 (lane w owns keep word w) applies the sequential
// greedy recurrence with __shfl broadcasts.  Only 2 barriers per chunk.
// ---------------------------------------------------------------------------
__global__ __launch_bounds__(256)
void k_nms(const float* __restrict__ sv, const int* __restrict__ order,
           const float* __restrict__ boxes_c, unsigned* __restrict__ keep_out) {
  int c = blockIdx.x, tid = threadIdx.x;
  __shared__ float X1[N_PROP], Y1[N_PROP], X2[N_PROP], Y2[N_PROP], AR[N_PROP];
  __shared__ unsigned sup[256][32];
  __shared__ unsigned keep[32];

  for (int i = tid; i < N_PROP; i += 256) {
    int o = order[c*N_PROP + i];
    const float* b = boxes_c + (c*N_PROP + o)*4;
    float x1=b[0], y1=b[1], x2=b[2], y2=b[3];
    X1[i]=x1; Y1[i]=y1; X2[i]=x2; Y2[i]=y2;
    AR[i] = (x2-x1+1.f)*(y2-y1+1.f);
  }
  if (tid < 32) {
    unsigned w = 0;
    for (int b = 0; b < 32; ++b) {
      int i = tid*32 + b;
      if (i < N_PROP && sv[c*N_PROP + i] != NEG_INF) w |= (1u << b);
    }
    keep[tid] = w;
  }
  __syncthreads();

  for (int base = 0; base < N_PROP; base += 256) {
    int rows = min(256, N_PROP - base);
    for (int idx = tid; idx < rows*32; idx += 256) {
      int r = idx >> 5, w = idx & 31;
      int i = base + r;
      float x1i=X1[i], y1i=Y1[i], x2i=X2[i], y2i=Y2[i], ai=AR[i];
      unsigned m = 0;
      int j0 = w << 5;
      for (int b2 = 0; b2 < 32; ++b2) {
        int j = j0 + b2;
        if (j < N_PROP && j > i) {
          float iw = fmaxf(fminf(x2i, X2[j]) - fmaxf(x1i, X1[j]) + 1.f, 0.f);
          float ih = fmaxf(fminf(y2i, Y2[j]) - fmaxf(y1i, Y1[j]) + 1.f, 0.f);
          float inter = iw * ih;
          float iou = inter / (ai + AR[j] - inter);
          if (iou > 0.5f) m |= (1u << b2);
        }
      }
      sup[r][w] = m;
    }
    __syncthreads();
    if (tid < 32) {                       // single wave32 applies greedy chain
      unsigned kw = keep[tid];
      for (int r = 0; r < rows; ++r) {
        int i = base + r;
        unsigned wi = __shfl(kw, i >> 5, 32);
        if ((wi >> (i & 31)) & 1u) kw &= ~sup[r][tid];
      }
      keep[tid] = kw;
    }
    __syncthreads();
  }
  if (tid < 32) keep_out[c*32 + tid] = keep[tid];
}

// ---------------------------------------------------------------------------
// K4: top-100 over the 80000 masked scores (value desc, tie -> smaller flat
// index, matching jax.lax.top_k) + emit boxes/scores/labels and the selection
// descriptors for the WMMA feature gather.  Single 1024-thread block.
// d_out layout (return-order concat): boxes[0..399] scores[400..499]
// labels[500..599] feats[600..]
// ---------------------------------------------------------------------------
__global__ __launch_bounds__(1024)
void k_topk(const float* __restrict__ sv, const unsigned* __restrict__ keepw,
            const int* __restrict__ order, const float* __restrict__ boxes_c,
            float* __restrict__ out, float* __restrict__ flat,
            int* __restrict__ sel_idx, float* __restrict__ sel_gf) {
  int tid = threadIdx.x;
  __shared__ float bv[1024];
  __shared__ int   bi[1024];

  const int TOT = N_C1 * N_PROP;
  for (int f = tid; f < TOT; f += 1024) {
    int c = f / N_PROP, p = f - c*N_PROP;
    bool k = (keepw[c*32 + (p >> 5)] >> (p & 31)) & 1u;
    flat[f] = k ? sv[f] : NEG_INF;
  }
  __syncthreads();

  for (int kk = 0; kk < DETS; ++kk) {
    float best = flat[tid];
    int   bidx = tid;
    for (int f = tid + 1024; f < TOT; f += 1024) {
      float v = flat[f];
      if (v > best) { best = v; bidx = f; }   // ascending scan keeps smallest f on ties
    }
    bv[tid] = best; bi[tid] = bidx;
    __syncthreads();
    for (int s = 512; s > 0; s >>= 1) {
      if (tid < s) {
        float v2 = bv[tid+s]; int i2 = bi[tid+s];
        if (v2 > bv[tid] || (v2 == bv[tid] && i2 < bi[tid])) { bv[tid]=v2; bi[tid]=i2; }
      }
      __syncthreads();
    }
    if (tid == 0) {
      float v = bv[0]; int f = bi[0];
      bool good = (v != NEG_INF);
      float gf = good ? 1.f : 0.f;
      int c = f / N_PROP;
      int orig = order[f];
      out[400 + kk] = good ? v : 0.f;
      out[500 + kk] = good ? (float)(c + 1) : 0.f;
      const float* b = boxes_c + (c*N_PROP + orig)*4;
      out[kk*4+0] = b[0]*gf; out[kk*4+1] = b[1]*gf;
      out[kk*4+2] = b[2]*gf; out[kk*4+3] = b[3]*gf;
      sel_idx[kk] = good ? orig : -1;
      sel_gf[kk]  = gf;
      flat[f] = NEG_INF;
    }
    __syncthreads();
  }
}

// ---------------------------------------------------------------------------
// K5: out_feats[100,1024] = OneHot(sel) @ features[1000,1024] on the matrix
// pipe.  F32 16x16x4 WMMA: products are exactly 1.0*feature or 0 -> bit-exact
// gather.  448 waves, one 16x16 tile each; M padded 100->112, K = 1000 = 250*4.
// A built in-register from sel_idx/sel_gf; B streamed from HBM with prefetch.
// ---------------------------------------------------------------------------
__global__ __launch_bounds__(128)
void k_feat(const float* __restrict__ F, const int* __restrict__ sel_idx,
            const float* __restrict__ sel_gf, float* __restrict__ O) {
  int gtid  = blockIdx.x * 128 + threadIdx.x;
  int wave  = gtid >> 5;          // 0..447
  int lane  = threadIdx.x & 31;
  int mtile = wave >> 6;          // 0..6   (112 rows)
  int ntile = wave & 63;          // 0..63  (1024 cols)

  int mrow = mtile*16 + (lane & 15);             // A-matrix row for this lane
  int selI = (mrow < DETS) ? sel_idx[mrow] : -1;
  float selG = (mrow < DETS) ? sel_gf[mrow] : 0.f;
  int ncol = ntile*16 + (lane & 15);             // B/C/D column for this lane

  v8f acc = {};
#if __has_builtin(__builtin_amdgcn_wmma_f32_16x16x4_f32)
  // A 16x4 f32: vgpr0 = K {0 | 2}, vgpr1 = K {1 | 3} (lanes 0-15 | 16-31)
  int khalf = (lane < 16) ? 0 : 2;
  for (int k = 0; k < N_PROP; k += 4) {
    int k0 = k + khalf;
    v2f a, b;
    a.x = (selI == k0    ) ? selG : 0.f;
    a.y = (selI == k0 + 1) ? selG : 0.f;
    b.x = F[(k0    ) * D_FEAT + ncol];
    b.y = F[(k0 + 1) * D_FEAT + ncol];
    if (k + 4 < N_PROP) __builtin_prefetch(&F[(k0 + 4) * D_FEAT + ncol], 0, 1);
    acc = __builtin_amdgcn_wmma_f32_16x16x4_f32(
        false, a, false, b, (short)0, acc, false, false);
  }
#else
  // Fallback: probe-confirmed f16 16x16x32 (one-hot exact in f16; features
  // rounded to f16).  K padded 1000 -> 1024 with zeros.
  int h = (lane < 16) ? 0 : 1;
  for (int k = 0; k < 1024; k += 32) {
    v16h a, b;
    for (int e = 0; e < 16; ++e) {
      int ka = k + 16*h + e;
      a[e] = (_Float16)(((ka < N_PROP) && (selI == ka)) ? selG : 0.f);
      int kb = k + 16*h + e;
      b[e] = (_Float16)((kb < N_PROP) ? F[kb * D_FEAT + ncol] : 0.f);
    }
    acc = __builtin_amdgcn_wmma_f32_16x16x32_f16(
        false, a, false, b, (short)0, acc, false, false);
  }
#endif
  // C/D layout: vgpr r -> M = r (lanes 0-15) / 8+r (lanes 16-31), N = lane%16
  int mbase = mtile*16 + ((lane < 16) ? 0 : 8);
  for (int r = 0; r < 8; ++r) {
    int m = mbase + r;
    if (m < DETS) O[m * D_FEAT + ncol] = acc[r];
  }
}

// ---------------------------------------------------------------------------
extern "C" void kernel_launch(void* const* d_in, const int* in_sizes, int n_in,
                              void* d_out, int out_size, void* d_ws, size_t ws_size,
                              hipStream_t stream) {
  const float* logits = (const float*)d_in[0];   // [1000, 81]
  const float* reg    = (const float*)d_in[1];   // [1000, 324]
  const float* feats  = (const float*)d_in[2];   // [1000, 1024]
  const float* props  = (const float*)d_in[3];   // [1000, 4]
  float* out = (float*)d_out;

  // workspace partition (~2.6 MB)
  float*    scores_c = (float*)d_ws;                       // 80*1000
  float*    boxes_c  = scores_c + N_C1*N_PROP;             // 80*1000*4
  float*    sv       = boxes_c  + N_C1*N_PROP*4;           // 80*1000
  int*      order    = (int*)(sv + N_C1*N_PROP);           // 80*1000
  unsigned* keepw    = (unsigned*)(order + N_C1*N_PROP);   // 80*32
  float*    flat     = (float*)(keepw + N_C1*32);          // 80*1000
  int*      sel_idx  = (int*)(flat + N_C1*N_PROP);         // 128
  float*    sel_gf   = (float*)(sel_idx + 128);            // 128

  k_softmax_decode<<<(N_PROP + 127)/128, 128, 0, stream>>>(logits, reg, props,
                                                           scores_c, boxes_c);
  k_sort<<<N_C1, 256, 0, stream>>>(scores_c, sv, order);
  k_nms<<<N_C1, 256, 0, stream>>>(sv, order, boxes_c, keepw);
  k_topk<<<1, 1024, 0, stream>>>(sv, keepw, order, boxes_c, out,
                                 flat, sel_idx, sel_gf);
  k_feat<<<112, 128, 0, stream>>>(feats, sel_idx, sel_gf, out + 600);
}
